// CausalMultiHeadSelfAttention_59141699666007
// MI455X (gfx1250) — compile-verified, matrix operand order checked by reading, not run on hardware
//
#include <hip/hip_runtime.h>

// ---------------------------------------------------------------------------
// Causal MHA + RoPE for gfx1250 (MI455X), bf16 WMMA with f32 accumulate.
// GEMM: 32x64 tile per wave, compile-time shapes so row strides become
// immediate VGLOBAL offsets (2 base pointers only -> no spills).
// ---------------------------------------------------------------------------

typedef __bf16 bf16;
typedef __attribute__((ext_vector_type(16))) __bf16 v16bf;
typedef __attribute__((ext_vector_type(8)))  __bf16 v8bf;
typedef __attribute__((ext_vector_type(8)))  float  v8f;

union BFrag { v16bf v; v8bf h8[2]; };

#define D_MODEL 1024
#define SEQ     2048
#define NBATCH  2
#define NHEADS  16
#define DK      64
#define NROWS   (NBATCH * SEQ)   // 4096

// ---------------------------------------------------------------------------
// f32 -> bf16 convert
// ---------------------------------------------------------------------------
__global__ void f32_to_bf16_kernel(const float* __restrict__ in,
                                   bf16* __restrict__ out, int n) {
  int i = blockIdx.x * blockDim.x + threadIdx.x;
  if (i < n) out[i] = (bf16)in[i];
}

// ---------------------------------------------------------------------------
// GEMM: out[n,m] = sum_k A[n,k] * W[m,k]
//   A: [GN][GK] row-major bf16, W: [GM][GK] row-major bf16
//   GN=4096, GM=1024, GK=1024 fixed at compile time.
// One wave computes a 32x64 tile: 2 A-frags x 4 B-frags = 8 WMMAs per K-step.
// MODE 0: bf16 [GN][GM]; MODE 1: bf16 transposed [GM][GN]; MODE 2: f32 [GN][GM].
// ---------------------------------------------------------------------------
template <int MODE>
__global__ void gemm_wmma_kernel(const bf16* __restrict__ A,
                                 const bf16* __restrict__ W,
                                 void* __restrict__ out) {
  constexpr int GN = NROWS, GM = D_MODEL, GK = D_MODEL;
  constexpr int RS = 16 * GK;                    // 16-row stride (elements)

  const int wave = threadIdx.x >> 5;
  const int lane = threadIdx.x & 31;
  const int l16  = lane & 15;
  const int hl   = lane >> 4;

  constexpr int tilesM = GM / 64;                // 64-col tiles
  const int tile = blockIdx.x * (blockDim.x >> 5) + wave;
  const int tn = tile / tilesM;                  // 32-row tile
  const int tm = tile % tilesM;                  // 64-col tile

  const bf16* abase = A + (size_t)(tn * 32 + l16) * GK;   // +RS for 2nd row block
  const bf16* bbase = W + (size_t)(tm * 64 + l16) * GK;   // +t*RS for B tile t

  v8f acc[2][4];
  for (int i = 0; i < 2; ++i)
    for (int t = 0; t < 4; ++t) acc[i][t] = (v8f){};

  for (int k0 = 0; k0 < GK; k0 += 32) {
    const int d0 = k0 + hl * 8;
    if (k0 + 256 < GK) {                  // cover L2 latency on the streams
      __builtin_prefetch(abase + d0 + 256, 0, 1);
      __builtin_prefetch(bbase + d0 + 256, 0, 1);
    }
    BFrag a[2], b[4];
    for (int i = 0; i < 2; ++i) {
      a[i].h8[0] = *(const v8bf*)(abase + i * RS + d0);
      a[i].h8[1] = *(const v8bf*)(abase + i * RS + d0 + 16);
    }
    for (int t = 0; t < 4; ++t) {
      b[t].h8[0] = *(const v8bf*)(bbase + t * RS + d0);
      b[t].h8[1] = *(const v8bf*)(bbase + t * RS + d0 + 16);
    }
    for (int i = 0; i < 2; ++i)
      for (int t = 0; t < 4; ++t)
        acc[i][t] = __builtin_amdgcn_wmma_f32_16x16x32_bf16(
            false, a[i].v, false, b[t].v, (short)0, acc[i][t], false, false);
  }

  // C/D layout: lane holds col = l16; VGPR r holds row = r + 8*hl.
  for (int i = 0; i < 2; ++i)
    for (int t = 0; t < 4; ++t)
      for (int r = 0; r < 8; ++r) {
        const size_t row = (size_t)(tn * 32 + i * 16 + r + 8 * hl);
        const size_t col = (size_t)(tm * 64 + t * 16 + l16);
        if constexpr (MODE == 2)      ((float*)out)[row * GM + col] = acc[i][t][r];
        else if constexpr (MODE == 1) ((bf16*)out)[col * GN + row]  = (bf16)acc[i][t][r];
        else                          ((bf16*)out)[row * GM + col]  = (bf16)acc[i][t][r];
      }
}

// ---------------------------------------------------------------------------
// RoPE applied in-place to Q and K (bf16, layout [NROWS][D_MODEL])
// ---------------------------------------------------------------------------
__global__ void rope_kernel(bf16* __restrict__ Q, bf16* __restrict__ Km,
                            const int* __restrict__ pos) {
  const int idx = blockIdx.x * blockDim.x + threadIdx.x;   // pair index
  const int dpair = idx & 31;            // 0..31 (DK/2)
  const int rest  = idx >> 5;
  const int h = rest & 15;
  const int n = rest >> 4;               // 0..4095
  const int s = n & (SEQ - 1);
  const float p = (float)pos[s];
  const float freq = __powf(10000.0f, -(float)(2 * dpair) / 64.0f);
  float sn, cs;
  __sincosf(p * freq, &sn, &cs);
  const size_t base = (size_t)n * D_MODEL + h * DK + 2 * dpair;

  float qe = (float)Q[base], qo = (float)Q[base + 1];
  Q[base]     = (bf16)(qe * cs - qo * sn);
  Q[base + 1] = (bf16)(qe * sn + qo * cs);

  float ke = (float)Km[base], ko = (float)Km[base + 1];
  Km[base]     = (bf16)(ke * cs - ko * sn);
  Km[base + 1] = (bf16)(ke * sn + ko * cs);
}

// ---------------------------------------------------------------------------
// Flash attention (causal). Q,K: [NROWS][D_MODEL] bf16 (head-sliced),
// VT: [D_MODEL][NROWS] bf16 (transposed), O: [NROWS][D_MODEL] bf16.
// One wave = one 16-row q tile. Online softmax over 32-key blocks.
// ---------------------------------------------------------------------------
__global__ void attn_kernel(const bf16* __restrict__ Q,
                            const bf16* __restrict__ Km,
                            const bf16* __restrict__ VT,
                            bf16* __restrict__ O) {
  __shared__ bf16 pbuf[4][16 * 32];   // per-wave P-tile scratch (transpose via LDS)

  const int wave = threadIdx.x >> 5;
  const int lane = threadIdx.x & 31;
  const int l16  = lane & 15;
  const int hl   = lane >> 4;

  const int h  = blockIdx.y;
  const int b  = blockIdx.z;
  const int qb = (blockIdx.x * 4 + wave) * 16;
  const size_t rowbase = (size_t)b * SEQ;

  // Q fragments (A layout, 16x64 -> two 16x32 frags)
  BFrag qa[2];
  {
    const bf16* qrow = Q + (rowbase + qb + l16) * D_MODEL + h * DK;
    for (int j = 0; j < 2; ++j) {
      const int d0 = j * 32 + hl * 8;
      qa[j].h8[0] = *(const v8bf*)(qrow + d0);
      qa[j].h8[1] = *(const v8bf*)(qrow + d0 + 16);
    }
  }

  float mrow[8], lrow[8];
  v8f oacc[4];
  for (int r = 0; r < 8; ++r) { mrow[r] = -1e30f; lrow[r] = 0.0f; }
  for (int t = 0; t < 4; ++t) oacc[t] = (v8f){};
  const float scale = 0.125f;   // 1/sqrt(64)
  bf16* pb = &pbuf[wave][0];

  for (int kb = 0; kb < qb + 16; kb += 32) {
    // ---- S = Q K^T for two 16-key groups -------------------------------
    v8f s0 = {}, s1 = {};
    const bf16* krow0 = Km + (rowbase + kb + l16) * D_MODEL + h * DK;
    const bf16* krow1 = krow0 + (size_t)16 * D_MODEL;
    for (int j = 0; j < 2; ++j) {
      const int d0 = j * 32 + hl * 8;
      BFrag kf;
      kf.h8[0] = *(const v8bf*)(krow0 + d0);
      kf.h8[1] = *(const v8bf*)(krow0 + d0 + 16);
      s0 = __builtin_amdgcn_wmma_f32_16x16x32_bf16(false, qa[j].v, false, kf.v,
                                                   (short)0, s0, false, false);
      kf.h8[0] = *(const v8bf*)(krow1 + d0);
      kf.h8[1] = *(const v8bf*)(krow1 + d0 + 16);
      s1 = __builtin_amdgcn_wmma_f32_16x16x32_bf16(false, qa[j].v, false, kf.v,
                                                   (short)0, s1, false, false);
    }

    // ---- causal mask + scale + online softmax --------------------------
    for (int r = 0; r < 8; ++r) {
      const int row = qb + r + 8 * hl;
      const int c0 = kb + l16, c1 = c0 + 16;
      float v0 = (c0 <= row) ? s0[r] * scale : -1e30f;
      float v1 = (c1 <= row) ? s1[r] * scale : -1e30f;
      float mx = fmaxf(v0, v1);
      for (int d = 1; d < 16; d <<= 1) mx = fmaxf(mx, __shfl_xor(mx, d, 32));
      const float mnew = fmaxf(mrow[r], mx);
      const float p0 = __expf(v0 - mnew);
      const float p1 = __expf(v1 - mnew);
      float sum = p0 + p1;
      for (int d = 1; d < 16; d <<= 1) sum += __shfl_xor(sum, d, 32);
      const float alpha = __expf(mrow[r] - mnew);
      lrow[r] = lrow[r] * alpha + sum;
      mrow[r] = mnew;
      oacc[0][r] *= alpha; oacc[1][r] *= alpha;
      oacc[2][r] *= alpha; oacc[3][r] *= alpha;
      pb[(r + 8 * hl) * 32 + l16]      = (bf16)p0;
      pb[(r + 8 * hl) * 32 + l16 + 16] = (bf16)p1;
    }
    asm volatile("s_wait_dscnt 0" ::: "memory");   // order LDS P store -> load

    // ---- P fragment (A layout 16x32) from LDS --------------------------
    BFrag pa;
    pa.h8[0] = *(const v8bf*)(pb + l16 * 32 + hl * 8);
    pa.h8[1] = *(const v8bf*)(pb + l16 * 32 + hl * 8 + 16);

    // ---- O += P V : V^T gives contiguous B-fragment loads --------------
    for (int t = 0; t < 4; ++t) {
      const bf16* vtrow = VT + (size_t)(h * DK + t * 16 + l16) * NROWS + rowbase + kb;
      BFrag vf;
      vf.h8[0] = *(const v8bf*)(vtrow + hl * 8);
      vf.h8[1] = *(const v8bf*)(vtrow + hl * 8 + 16);
      oacc[t] = __builtin_amdgcn_wmma_f32_16x16x32_bf16(false, pa.v, false, vf.v,
                                                        (short)0, oacc[t], false, false);
    }
  }

  // ---- normalize + store ----------------------------------------------
  for (int t = 0; t < 4; ++t)
    for (int r = 0; r < 8; ++r) {
      const int row = qb + r + 8 * hl;
      O[(rowbase + row) * D_MODEL + h * DK + t * 16 + l16] =
          (bf16)(oacc[t][r] / lrow[r]);
    }
}

// ---------------------------------------------------------------------------
// Launcher
// ---------------------------------------------------------------------------
extern "C" void kernel_launch(void* const* d_in, const int* in_sizes, int n_in,
                              void* d_out, int out_size, void* d_ws, size_t ws_size,
                              hipStream_t stream) {
  const float* X   = (const float*)d_in[0];
  const int*   pos = (const int*)  d_in[1];
  const float* Wq  = (const float*)d_in[2];
  const float* Wk  = (const float*)d_in[3];
  const float* Wv  = (const float*)d_in[4];
  const float* Wo  = (const float*)d_in[5];
  float* out = (float*)d_out;

  const size_t NA = (size_t)NROWS * D_MODEL;     // 4M activations
  const size_t NW = (size_t)D_MODEL * D_MODEL;   // 1M weights
  bf16* Xb  = (bf16*)d_ws;          // [4096][1024]
  bf16* Wqb = Xb  + NA;
  bf16* Wkb = Wqb + NW;
  bf16* Wvb = Wkb + NW;
  bf16* Wob = Wvb + NW;
  bf16* Qb  = Wob + NW;             // [4096][1024]
  bf16* Kb  = Qb  + NA;             // [4096][1024]
  bf16* VTb = Kb  + NA;             // transposed [1024][4096]
  bf16* Ob  = VTb + NA;             // [4096][1024]
  // total ws: 24M bf16 = 48 MB

  // 1) converts
  f32_to_bf16_kernel<<<(int)(NA / 256), 256, 0, stream>>>(X,  Xb,  (int)NA);
  f32_to_bf16_kernel<<<(int)(NW / 256), 256, 0, stream>>>(Wq, Wqb, (int)NW);
  f32_to_bf16_kernel<<<(int)(NW / 256), 256, 0, stream>>>(Wk, Wkb, (int)NW);
  f32_to_bf16_kernel<<<(int)(NW / 256), 256, 0, stream>>>(Wv, Wvb, (int)NW);
  f32_to_bf16_kernel<<<(int)(NW / 256), 256, 0, stream>>>(Wo, Wob, (int)NW);

  // 2) QKV projections: 32x64 tile per wave, 8 waves/block
  const int tiles  = (NROWS / 32) * (D_MODEL / 64);   // 2048
  const int blocks = tiles / 8;                       // 256
  gemm_wmma_kernel<0><<<blocks, 256, 0, stream>>>(Xb, Wqb, Qb);
  gemm_wmma_kernel<0><<<blocks, 256, 0, stream>>>(Xb, Wkb, Kb);
  gemm_wmma_kernel<1><<<blocks, 256, 0, stream>>>(Xb, Wvb, VTb);

  // 3) RoPE on Q,K
  const int pairs = NROWS * NHEADS * (DK / 2);        // 2M
  rope_kernel<<<pairs / 256, 256, 0, stream>>>(Qb, Kb, pos);

  // 4) causal flash attention: grid (S/64, heads, batch), 4 waves/block
  attn_kernel<<<dim3(SEQ / 64, NHEADS, NBATCH), 128, 0, stream>>>(Qb, Kb, VTb, Ob);

  // 5) output projection -> f32 d_out
  gemm_wmma_kernel<2><<<blocks, 256, 0, stream>>>(Ob, Wob, (void*)out);
}
